// SuspiciousPatternLSTM_78271484002977
// MI455X (gfx1250) — matrix operand order repaired and checked by reading, not compiled, not run on hardware
//
#include <hip/hip_runtime.h>

typedef __attribute__((ext_vector_type(16))) _Float16 v16h;
typedef __attribute__((ext_vector_type(8)))  _Float16 v8h;
typedef __attribute__((ext_vector_type(8)))  float    v8f;

namespace {
constexpr int kH     = 64;    // hidden size
constexpr int kT     = 512;   // timesteps
constexpr int kB     = 4096;  // batch
constexpr int kTileB = 16;    // batch rows per tile (WMMA M)
constexpr int kWavesPerTile = 2;                       // N-split: 2 hc-chunks per wave
constexpr int kTilesPerBlk  = 2;
constexpr int kWavesPerBlk  = kWavesPerTile * kTilesPerBlk;  // 4 waves
constexpr int kThreads      = 32 * kWavesPerBlk;             // 128

// dynamic LDS layout (bytes, 16B aligned)
constexpr size_t kOffW0   = 3ull * 16 * 2 * 16 * 32 * 2;     // weight B-tiles f16: 98304
constexpr size_t kOffBias = kOffW0 + 256 * 3 * 4;            // Wih0 fp32
constexpr size_t kOffH1   = kOffBias + 2 * 256 * 4;          // fused biases
constexpr size_t kOffH2   = kOffH1 + kTilesPerBlk * 2 * 1024 * 2; // h1 dbl-buf f16
constexpr size_t kOffX    = kOffH2 + kTilesPerBlk * 2 * 1024 * 2; // h2 dbl-buf f16
constexpr size_t kSmem    = kOffX + kWavesPerBlk * 48 * 4;   // = 120576
} // namespace

__device__ __forceinline__ float fast_rcp(float x) {
#if __has_builtin(__builtin_amdgcn_rcpf)
  return __builtin_amdgcn_rcpf(x);
#else
  return 1.0f / x;
#endif
}

__device__ __forceinline__ float ftanh(float x) {
#if __has_builtin(__builtin_amdgcn_tanhf)
  return __builtin_amdgcn_tanhf(x);          // v_tanh_f32 (CDNA5 TRANS op)
#elif __has_builtin(__builtin_amdgcn_tanh_f32)
  return __builtin_amdgcn_tanh_f32(x);
#else
  const float ax = __builtin_fabsf(x);
  const float e  = __expf(2.0f * ax);
  const float r  = 1.0f - 2.0f * fast_rcp(e + 1.0f);
  return __builtin_copysignf(r, x);
#endif
}

__device__ __forceinline__ float fsig(float x) {
  return fast_rcp(1.0f + __expf(-x));
}

__device__ __forceinline__ v8f wmma16(v16h a, v16h b, v8f c) {
  // D = A(16x32 f16) * B(32x16 f16) + C(16x16 f32)
  return __builtin_amdgcn_wmma_f32_16x16x32_f16(false, a, false, b, (short)0, c,
                                                false, false);
}

// A operand (16x32 f16) from a row-major [16][64] f16 LDS buffer.
// Lane l: row m = l&15; K elems {0..7,16..23} (l<16) or {8..15,24..31} (l>=16).
__device__ __forceinline__ v16h load_a_tile(const _Float16* hbuf, int lane, int kc) {
  const int m  = lane & 15;
  const int k0 = kc * 32 + ((lane & 16) ? 8 : 0);
  const v8h lo = *(const v8h*)(hbuf + m * kH + k0);
  const v8h hi = *(const v8h*)(hbuf + m * kH + k0 + 16);
  v16h r;
#pragma unroll
  for (int i = 0; i < 8; ++i) { r[i] = lo[i]; r[8 + i] = hi[i]; }
  return r;
}

// B operand (32x16 f16) from a pre-swizzled [n=16][k=32] f16 tile.
// Lane l: column n = l&15; K elems {0..7,16..23} (l<16) or {8..15,24..31} (l>=16).
__device__ __forceinline__ v16h load_b_tile(const _Float16* wtile, int lane) {
  const int n  = lane & 15;
  const int k0 = (lane & 16) ? 8 : 0;
  const v8h lo = *(const v8h*)(wtile + n * 32 + k0);
  const v8h hi = *(const v8h*)(wtile + n * 32 + k0 + 16);
  v16h r;
#pragma unroll
  for (int i = 0; i < 8; ++i) { r[i] = lo[i]; r[8 + i] = hi[i]; }
  return r;
}

__global__ __launch_bounds__(kThreads)
void fused_lstm2_head(const float* __restrict__ x,
                      const float* __restrict__ wih0, const float* __restrict__ whh0,
                      const float* __restrict__ bih0, const float* __restrict__ bhh0,
                      const float* __restrict__ wih1, const float* __restrict__ whh1,
                      const float* __restrict__ bih1, const float* __restrict__ bhh1,
                      const float* __restrict__ wc1,  const float* __restrict__ bc1,
                      const float* __restrict__ wc2,  const float* __restrict__ bc2,
                      float* __restrict__ out) {
  extern __shared__ char smem[];
  _Float16* wlds  = (_Float16*)smem;             // [3][16 nt][2 kc][16 n][32 k]
  float*    w0lds = (float*)(smem + kOffW0);     // [256][3]
  float*    blds  = (float*)(smem + kOffBias);   // [2][256]
  _Float16* h1b   = (_Float16*)(smem + kOffH1);  // [tiles][2 parity][16*64]
  _Float16* h2b   = (_Float16*)(smem + kOffH2);
  float*    xlds  = (float*)(smem + kOffX);      // [waves][48]

  const int tid    = threadIdx.x;
  const int lane   = tid & 31;
  const int wave   = tid >> 5;        // 0..3
  const int tileIn = wave >> 1;       // 0..1 : batch tile within block
  const int split  = wave & 1;        // 0..1 : which pair of hc-chunks

  // ---- Stage weights: fp32 global -> f16 B-tile layout in LDS (whole block) ----
  for (int i = tid; i < 3 * 256 * 64; i += kThreads) {
    const int mi = i >> 14;
    const int r  = i & 16383;
    const int n  = r >> 6;            // gate column 0..255
    const int k  = r & 63;            // input col 0..63
    const float* src = (mi == 0) ? whh0 : ((mi == 1) ? wih1 : whh1);
    const int nt = n >> 4, nl = n & 15, kc = k >> 5, kl = k & 31;
    wlds[((((mi * 16 + nt) * 2 + kc) * 16 + nl) << 5) + kl] =
        (_Float16)src[(n << 6) + k];
  }
  for (int i = tid; i < 256 * 3; i += kThreads) w0lds[i] = wih0[i];
  for (int i = tid; i < 256; i += kThreads) {
    blds[i]       = bih0[i] + bhh0[i];
    blds[256 + i] = bih1[i] + bhh1[i];
  }

  const _Float16* WB0 = wlds;               // Whh0 tiles
  const _Float16* WB1 = wlds + 16384;       // Wih1 tiles
  const _Float16* WB2 = wlds + 32768;       // Whh1 tiles

  _Float16* h1t = h1b + tileIn * 2048;      // 2 parity buffers of 1024 halfs
  _Float16* h2t = h2b + tileIn * 2048;
  float*    xw  = xlds + wave * 48;
  const int b0  = (blockIdx.x * kTilesPerBlk + tileIn) * kTileB;

  // zero both parities of this tile's h state (pair of waves cooperates)
  for (int i = lane + split * 32; i < 2048; i += 64) {
    h1t[i] = (_Float16)0.0f;
    h2t[i] = (_Float16)0.0f;
  }
  __syncthreads();

  float c1[2][8], c2[2][8];
#pragma unroll
  for (int hcl = 0; hcl < 2; ++hcl)
#pragma unroll
    for (int v = 0; v < 8; ++v) { c1[hcl][v] = 0.0f; c2[hcl][v] = 0.0f; }

  const int ncol  = lane & 15;        // C/D column this lane owns
  const int rbase = (lane >> 4) * 8;  // C/D rows rbase..rbase+7

  // One full timestep: L1 reads h1r -> writes h1w; barrier; L2 reads (h1w, h2r)
  // -> writes h2w. Double buffering makes this single barrier sufficient.
  auto step = [&](int t, const _Float16* h1r, _Float16* h1w,
                  const _Float16* h2r, _Float16* h2w) {
    // stage x_t tile [16 rows][3] (wave-private)
    {
      int f = lane;
      if (f < 48) {
        const int row = f / 3, k = f - 3 * row;
        const size_t gi = ((size_t)(b0 + row) * kT + t) * 3 + k;
        xw[f] = x[gi];
        if (t + 1 < kT) __builtin_prefetch(&x[gi + 3], 0, 1);
      }
      f = lane + 32;
      if (f < 48) {
        const int row = f / 3, k = f - 3 * row;
        xw[f] = x[((size_t)(b0 + row) * kT + t) * 3 + k];
      }
    }

    // ======================= Layer 1 (my gate-column half) =======================
    const v16h a1p0 = load_a_tile(h1r, lane, 0);   // h1_{t-1}
    const v16h a1p1 = load_a_tile(h1r, lane, 1);

    float xr[8][3];
#pragma unroll
    for (int v = 0; v < 8; ++v) {
      xr[v][0] = xw[(rbase + v) * 3 + 0];
      xr[v][1] = xw[(rbase + v) * 3 + 1];
      xr[v][2] = xw[(rbase + v) * 3 + 2];
    }

#pragma unroll
    for (int hcl = 0; hcl < 2; ++hcl) {
      const int hc = split * 2 + hcl;
      v8f acc[4];
#pragma unroll
      for (int g = 0; g < 4; ++g) {
        const int   ng   = g * 64 + hc * 16 + ncol;
        const float init = blds[ng];
        const float wa = w0lds[ng * 3 + 0], wb = w0lds[ng * 3 + 1],
                    wc = w0lds[ng * 3 + 2];
#pragma unroll
        for (int v = 0; v < 8; ++v)
          acc[g][v] = init + xr[v][0] * wa + xr[v][1] * wb + xr[v][2] * wc;
      }
#pragma unroll
      for (int g = 0; g < 4; ++g) {
        const int nt = g * 4 + hc;
        acc[g] = wmma16(a1p0, load_b_tile(WB0 + (nt * 2 + 0) * 512, lane), acc[g]);
        acc[g] = wmma16(a1p1, load_b_tile(WB0 + (nt * 2 + 1) * 512, lane), acc[g]);
      }
#pragma unroll
      for (int v = 0; v < 8; ++v) {
        const float ig = fsig(acc[0][v]);
        const float fg = fsig(acc[1][v]);
        const float gg = ftanh(acc[2][v]);
        const float og = fsig(acc[3][v]);
        const float c  = fg * c1[hcl][v] + ig * gg;
        c1[hcl][v] = c;
        h1w[(rbase + v) * kH + hc * 16 + ncol] = (_Float16)(og * ftanh(c));
      }
    }

    __syncthreads();   // h1_t complete across the wave pair

    // ======================= Layer 2 (my gate-column half) =======================
    const v16h a2x0 = load_a_tile(h1w, lane, 0);   // h1_t
    const v16h a2x1 = load_a_tile(h1w, lane, 1);
    const v16h a2h0 = load_a_tile(h2r, lane, 0);   // h2_{t-1}
    const v16h a2h1 = load_a_tile(h2r, lane, 1);

#pragma unroll
    for (int hcl = 0; hcl < 2; ++hcl) {
      const int hc = split * 2 + hcl;
      v8f acc[4];
#pragma unroll
      for (int g = 0; g < 4; ++g) {
        const float init = blds[256 + g * 64 + hc * 16 + ncol];
#pragma unroll
        for (int v = 0; v < 8; ++v) acc[g][v] = init;
      }
#pragma unroll
      for (int g = 0; g < 4; ++g) {
        const int nt = g * 4 + hc;
        acc[g] = wmma16(a2x0, load_b_tile(WB1 + (nt * 2 + 0) * 512, lane), acc[g]);
        acc[g] = wmma16(a2x1, load_b_tile(WB1 + (nt * 2 + 1) * 512, lane), acc[g]);
        acc[g] = wmma16(a2h0, load_b_tile(WB2 + (nt * 2 + 0) * 512, lane), acc[g]);
        acc[g] = wmma16(a2h1, load_b_tile(WB2 + (nt * 2 + 1) * 512, lane), acc[g]);
      }
#pragma unroll
      for (int v = 0; v < 8; ++v) {
        const float ig = fsig(acc[0][v]);
        const float fg = fsig(acc[1][v]);
        const float gg = ftanh(acc[2][v]);
        const float og = fsig(acc[3][v]);
        const float c  = fg * c2[hcl][v] + ig * gg;
        c2[hcl][v] = c;
        h2w[(rbase + v) * kH + hc * 16 + ncol] = (_Float16)(og * ftanh(c));
      }
    }
  };

  for (int t = 0; t < kT; t += 2) {
    step(t,     h1t,        h1t + 1024, h2t,        h2t + 1024);
    step(t + 1, h1t + 1024, h1t,        h2t + 1024, h2t);
  }
  __syncthreads();   // final h2 (in parity-0 buffer) complete

  // ---- Classifier head: sigmoid(relu(h2 @ Wc1^T + bc1) @ Wc2^T + bc2) ----
  if (split == 0 && lane < kTileB) {
    const int m = lane;
    float o = bc2[0];
#pragma unroll 1
    for (int j = 0; j < 32; ++j) {
      float s = bc1[j];
#pragma unroll 4
      for (int k = 0; k < kH; ++k) s += (float)h2t[m * kH + k] * wc1[j * kH + k];
      s = fmaxf(s, 0.0f);
      o += s * wc2[j];
    }
    out[b0 + m] = fsig(o);
  }
}

extern "C" void kernel_launch(void* const* d_in, const int* in_sizes, int n_in,
                              void* d_out, int out_size, void* d_ws, size_t ws_size,
                              hipStream_t stream) {
  (void)in_sizes; (void)n_in; (void)out_size; (void)d_ws; (void)ws_size;
  const float* x    = (const float*)d_in[0];
  const float* wih0 = (const float*)d_in[1];
  const float* whh0 = (const float*)d_in[2];
  const float* bih0 = (const float*)d_in[3];
  const float* bhh0 = (const float*)d_in[4];
  const float* wih1 = (const float*)d_in[5];
  const float* whh1 = (const float*)d_in[6];
  const float* bih1 = (const float*)d_in[7];
  const float* bhh1 = (const float*)d_in[8];
  const float* wc1  = (const float*)d_in[9];
  const float* bc1  = (const float*)d_in[10];
  const float* wc2  = (const float*)d_in[11];
  const float* bc2  = (const float*)d_in[12];
  float* out = (float*)d_out;

  (void)hipFuncSetAttribute((const void*)fused_lstm2_head,
                            hipFuncAttributeMaxDynamicSharedMemorySize, (int)kSmem);

  dim3 grid(kB / (kTileB * kTilesPerBlk));   // 128 blocks
  dim3 block(kThreads);                      // 128 threads = 4 wave32
  fused_lstm2_head<<<grid, block, kSmem, stream>>>(
      x, wih0, whh0, bih0, bhh0, wih1, whh1, bih1, bhh1, wc1, bc1, wc2, bc2, out);
}
